// CausalSelfAttention_7876970020932
// MI455X (gfx1250) — matrix operand order repaired and checked by reading, not compiled, not run on hardware
//
#include <hip/hip_runtime.h>
#include <math.h>

// ---------- types ----------
typedef __attribute__((ext_vector_type(16))) __bf16 v16bf;
typedef __attribute__((ext_vector_type(8)))  float  v8f;
typedef __attribute__((ext_vector_type(4)))  unsigned int v4u;
typedef __attribute__((ext_vector_type(4)))  int v4i;

union FragBF { v16bf v; v4u q[2]; };

// Async global->LDS path (CDNA5) if the toolchain exposes the builtins.
#if defined(__AMDGCN__) && __has_builtin(__builtin_amdgcn_global_load_async_to_lds_b128) && __has_builtin(__builtin_amdgcn_s_wait_asynccnt)
#define USE_ASYNC_LDS 1
typedef __attribute__((address_space(1))) v4i v4i_g;   // global
typedef __attribute__((address_space(3))) v4i v4i_l;   // LDS
#else
#define USE_ASYNC_LDS 0
#endif

// ---------- problem constants ----------
static constexpr int kB = 4;
static constexpr int kT = 2048;
static constexpr int kC = 2048;
static constexpr int kH = 16;
static constexpr int kD = 128;           // head dim
static constexpr int kM = kB * kT;       // 8192 rows for projections

// ---------- fp32 -> bf16 convert ----------
__global__ void cvt_f32_bf16(const float* __restrict__ src, void* __restrict__ dst_, int n) {
    __bf16* dst = (__bf16*)dst_;
    int i = blockIdx.x * blockDim.x + threadIdx.x;
    int stride = gridDim.x * blockDim.x;
    for (; i < n; i += stride) dst[i] = (__bf16)src[i];
}

// ---------- GEMM: Out[M,N] = A[M,K](bf16,row-major) @ W[N,K](bf16,row-major)^T ----------
// Block: 256 threads = 8 waves (2 M-waves x 4 N-waves); wave tile 64x64 -> block tile 128x256.
// Per K-step of 32: 8 fragment pairs loaded, 16 WMMA -> 4x register reuse per fragment.
template<bool OUT_BF16>
__global__ __launch_bounds__(256)
void gemm_bf16_wt(const void* __restrict__ A_, const void* __restrict__ W_,
                  void* __restrict__ Out_, int M, int N, int K) {
    const __bf16* A = (const __bf16*)A_;
    const __bf16* W = (const __bf16*)W_;
    const int lane = threadIdx.x & 31;
    const int wave = threadIdx.x >> 5;
    const int wm = wave & 1, wn = wave >> 1;
    const int m0 = blockIdx.y * 128 + wm * 64;
    const int n0 = blockIdx.x * 256 + wn * 64;
    const int grp = lane >> 4, l16 = lane & 15;

    v8f acc[4][4];
    for (int i = 0; i < 4; i++)
        for (int j = 0; j < 4; j++)
            for (int e = 0; e < 8; e++) acc[i][j][e] = 0.0f;

    // A frag (16-bit A 16x32 layout): elems 0..7 at k=grp*8+0..7, elems 8..15 at k=16+grp*8+0..7
    const __bf16* aRow[4];
    // B frag (16-bit B 32x16 layout): lane n = l16, elems 0..7 at k=grp*8+0..7, elems 8..15 at k=16+grp*8+0..7
    const __bf16* bRow[4];
    for (int i = 0; i < 4; i++) aRow[i] = A + (size_t)(m0 + i * 16 + l16) * K + grp * 8;
    for (int j = 0; j < 4; j++) bRow[j] = W + (size_t)(n0 + j * 16 + l16) * K + grp * 8;

    for (int k0 = 0; k0 < K; k0 += 32) {
        FragBF af[4], bf[4];
        for (int i = 0; i < 4; i++) {
            af[i].q[0] = *(const v4u*)(aRow[i] + k0);
            af[i].q[1] = *(const v4u*)(aRow[i] + k0 + 16);
        }
        for (int j = 0; j < 4; j++) {
            bf[j].q[0] = *(const v4u*)(bRow[j] + k0);
            bf[j].q[1] = *(const v4u*)(bRow[j] + k0 + 16);
        }
        __builtin_prefetch(aRow[0] + k0 + 128, 0, 1);
        __builtin_prefetch(bRow[0] + k0 + 128, 0, 1);
        for (int i = 0; i < 4; i++)
            for (int j = 0; j < 4; j++)
                acc[i][j] = __builtin_amdgcn_wmma_f32_16x16x32_bf16(
                    false, af[i].v, false, bf[j].v, (short)0, acc[i][j], false, false);
    }

    // C/D layout: lane n = l16 column; VGPR v -> row m = v + 8*grp
    for (int i = 0; i < 4; i++)
        for (int j = 0; j < 4; j++)
            for (int v = 0; v < 8; v++) {
                int m = m0 + i * 16 + v + grp * 8;
                int n = n0 + j * 16 + l16;
                float val = acc[i][j][v];
                if (OUT_BF16) ((__bf16*)Out_)[(size_t)m * N + n] = (__bf16)val;
                else          ((float*)Out_)[(size_t)m * N + n] = val;
            }
}

// ---------- RoPE in-place on bf16 q,k ----------
__global__ void rope_kernel(void* __restrict__ q_, void* __restrict__ k_,
                            const float* __restrict__ fcos, const float* __restrict__ fsin) {
    __bf16* q = (__bf16*)q_;
    __bf16* k = (__bf16*)k_;
    const int half = kD / 2;                       // 64
    const long n = (long)kB * kT * kH * half;      // one thread per rotation pair
    long i = (long)blockIdx.x * blockDim.x + threadIdx.x;
    if (i >= n) return;
    int d = (int)(i % half);
    int h = (int)((i / half) % kH);
    int t = (int)((i / ((long)half * kH)) % kT);
    int b = (int)(i / ((long)half * kH * kT));
    size_t base = ((size_t)(b * kT + t)) * kC + h * kD;
    float c1 = fcos[t * kD + d],        s1 = fsin[t * kD + d];
    float c2 = fcos[t * kD + d + half], s2 = fsin[t * kD + d + half];

    float q1 = (float)q[base + d], q2 = (float)q[base + d + half];
    q[base + d]        = (__bf16)(q1 * c1 - q2 * s1);
    q[base + d + half] = (__bf16)(q2 * c2 + q1 * s2);

    float k1 = (float)k[base + d], k2 = (float)k[base + d + half];
    k[base + d]        = (__bf16)(k1 * c1 - k2 * s1);
    k[base + d + half] = (__bf16)(k2 * c2 + k1 * s2);
}

// ---------- Flash attention (causal) ----------
// grid.x = b*H + h (64), grid.y = q-tile of 64 rows (32). 128 threads = 4 waves, 16 q-rows each.
__global__ __launch_bounds__(128)
void attn_kernel(const void* __restrict__ Q_, const void* __restrict__ K_,
                 const void* __restrict__ V_, void* __restrict__ Y_) {
    const __bf16* Q = (const __bf16*)Q_;
    const __bf16* Km = (const __bf16*)K_;
    const __bf16* Vm = (const __bf16*)V_;
    __bf16* Y = (__bf16*)Y_;

    const int bh = blockIdx.x;
    const int b = bh >> 4, h = bh & 15;
    const int qtile = blockIdx.y;
    const int wave = threadIdx.x >> 5;
    const int lane = threadIdx.x & 31;
    const int grp = lane >> 4, l16 = lane & 15;
    const int qrow0 = qtile * 64 + wave * 16;

    __shared__ alignas(16) __bf16 Klds[32][136];      // K tile, row-major [key][d]
    __shared__ alignas(16) __bf16 Vt[128][40];        // V tile transposed [d][key]
    __shared__ alignas(16) __bf16 Plds[4][16][40];    // per-wave P (16 x 32)

    // Load this wave's Q rows once: 4 K-chunks of 32 along D
    FragBF qf[4];
    {
        const __bf16* qrow = Q + ((size_t)(b * kT + qrow0 + l16)) * kC + h * kD + grp * 8;
        for (int dk = 0; dk < 4; dk++) {
            qf[dk].q[0] = *(const v4u*)(qrow + dk * 32);
            qf[dk].q[1] = *(const v4u*)(qrow + dk * 32 + 16);
        }
    }

    v8f O[8];
    for (int f = 0; f < 8; f++)
        for (int e = 0; e < 8; e++) O[f][e] = 0.0f;
    float mrun[8], lrun[8];
    for (int v = 0; v < 8; v++) { mrun[v] = -__builtin_inff(); lrun[v] = 0.0f; }

    const float scale = 0.08838834764831845f;   // 1/sqrt(128)
    const int ntiles = qtile * 2 + 2;           // key tiles of 32 up to causal limit

    for (int kt = 0; kt < ntiles; kt++) {
        const int t0 = kt * 32;
        // ---- cooperative staging: K row-major (async if available), V transposed ----
        {
            const int tl = threadIdx.x >> 2;          // 0..31 key row
            const int dc = (threadIdx.x & 3) * 32;    // 0,32,64,96
            const __bf16* krow = Km + ((size_t)(b * kT + t0 + tl)) * kC + h * kD + dc;
            const __bf16* vrow = Vm + ((size_t)(b * kT + t0 + tl)) * kC + h * kD + dc;
#if USE_ASYNC_LDS
            for (int q4 = 0; q4 < 4; q4++)
                __builtin_amdgcn_global_load_async_to_lds_b128(
                    (v4i_g*)(const v4i*)(krow + q4 * 8),
                    (v4i_l*)(v4i*)&Klds[tl][dc + q4 * 8],
                    0, 0);
#else
            for (int q4 = 0; q4 < 4; q4++)
                *(v4u*)&Klds[tl][dc + q4 * 8] = *(const v4u*)(krow + q4 * 8);
#endif
            for (int q4 = 0; q4 < 4; q4++) {
                union { v4u q; __bf16 e[8]; } u;
                u.q = *(const v4u*)(vrow + q4 * 8);
                for (int j = 0; j < 8; j++) Vt[dc + q4 * 8 + j][tl] = u.e[j];
            }
        }
#if USE_ASYNC_LDS
        __builtin_amdgcn_s_wait_asynccnt(0);
#endif
        __syncthreads();

        // ---- S = Q * K^T  (two 16-column chunks) ----
        v8f s[2];
        for (int c = 0; c < 2; c++) {
            for (int e = 0; e < 8; e++) s[c][e] = 0.0f;
            for (int dk = 0; dk < 4; dk++) {
                FragBF kf;
                const __bf16* kp = &Klds[c * 16 + l16][dk * 32 + grp * 8];
                kf.q[0] = *(const v4u*)kp;
                kf.q[1] = *(const v4u*)(kp + 16);
                s[c] = __builtin_amdgcn_wmma_f32_16x16x32_bf16(false, qf[dk].v, false, kf.v,
                                                               (short)0, s[c], false, false);
            }
        }

        // ---- scale + causal mask + online softmax ----
        for (int v = 0; v < 8; v++) {
            const int m = qrow0 + v + grp * 8;
            float x0 = s[0][v] * scale;
            float x1 = s[1][v] * scale;
            if (t0 + l16 > m)      x0 = -__builtin_inff();
            if (t0 + 16 + l16 > m) x1 = -__builtin_inff();
            float mx = fmaxf(x0, x1);
            mx = fmaxf(mx, __shfl_xor(mx, 1, 32));
            mx = fmaxf(mx, __shfl_xor(mx, 2, 32));
            mx = fmaxf(mx, __shfl_xor(mx, 4, 32));
            mx = fmaxf(mx, __shfl_xor(mx, 8, 32));
            float mnew = fmaxf(mrun[v], mx);
            float alpha = __expf(mrun[v] - mnew);
            float p0 = __expf(x0 - mnew);
            float p1 = __expf(x1 - mnew);
            float sum = p0 + p1;
            sum += __shfl_xor(sum, 1, 32);
            sum += __shfl_xor(sum, 2, 32);
            sum += __shfl_xor(sum, 4, 32);
            sum += __shfl_xor(sum, 8, 32);
            lrun[v] = lrun[v] * alpha + sum;
            mrun[v] = mnew;
            for (int f = 0; f < 8; f++) O[f][v] = O[f][v] * alpha;
            Plds[wave][v + grp * 8][l16]      = (__bf16)p0;
            Plds[wave][v + grp * 8][16 + l16] = (__bf16)p1;
        }
        __syncthreads();

        // ---- O += P * V  (A frag from Plds, B frags from transposed Vt) ----
        FragBF pf;
        {
            const __bf16* prow = &Plds[wave][l16][grp * 8];
            pf.q[0] = *(const v4u*)prow;
            pf.q[1] = *(const v4u*)(prow + 16);
        }
        for (int f = 0; f < 8; f++) {
            FragBF vf;
            const __bf16* vtr = &Vt[f * 16 + l16][grp * 8];
            vf.q[0] = *(const v4u*)vtr;
            vf.q[1] = *(const v4u*)(vtr + 16);
            O[f] = __builtin_amdgcn_wmma_f32_16x16x32_bf16(false, pf.v, false, vf.v,
                                                           (short)0, O[f], false, false);
        }
        __syncthreads();
    }

    // ---- normalize and store y (bf16) ----
    for (int v = 0; v < 8; v++) {
        float inv = 1.0f / lrun[v];
        size_t ybase = ((size_t)(b * kT + qrow0 + v + grp * 8)) * kC + h * kD;
        for (int f = 0; f < 8; f++)
            Y[ybase + f * 16 + l16] = (__bf16)(O[f][v] * inv);
    }
}

// ---------- host launch ----------
extern "C" void kernel_launch(void* const* d_in, const int* in_sizes, int n_in,
                              void* d_out, int out_size, void* d_ws, size_t ws_size,
                              hipStream_t stream) {
    (void)in_sizes; (void)n_in; (void)out_size; (void)ws_size;
    const float* x    = (const float*)d_in[0];
    const float* fcos = (const float*)d_in[1];
    const float* fsin = (const float*)d_in[2];
    const float* wq   = (const float*)d_in[3];
    const float* wk   = (const float*)d_in[4];
    const float* wv   = (const float*)d_in[5];
    const float* wo   = (const float*)d_in[6];

    char* ws = (char*)d_ws;
    const size_t szXB = (size_t)kM * kC * 2;   // 32MB (bf16 B*T*C)
    const size_t szWB = (size_t)kC * kC * 2;   // 8MB
    void* xb  = ws;
    void* wqb = ws + szXB;
    void* wkb = ws + szXB + szWB;
    void* wvb = ws + szXB + 2 * szWB;
    void* wob = ws + szXB + 3 * szWB;
    void* qb  = ws + szXB + 4 * szWB;
    void* kb  = ws + 2 * szXB + 4 * szWB;
    void* vb  = ws + 3 * szXB + 4 * szWB;
    void* yb  = ws + 4 * szXB + 4 * szWB;

    const int nX = kM * kC;       // 16.7M
    const int nW = kC * kC;       // 4.2M
    cvt_f32_bf16<<<4096, 256, 0, stream>>>(x,  xb,  nX);
    cvt_f32_bf16<<<2048, 256, 0, stream>>>(wq, wqb, nW);
    cvt_f32_bf16<<<2048, 256, 0, stream>>>(wk, wkb, nW);
    cvt_f32_bf16<<<2048, 256, 0, stream>>>(wv, wvb, nW);
    cvt_f32_bf16<<<2048, 256, 0, stream>>>(wo, wob, nW);

    dim3 gg(kC / 256, kM / 128);  // (8, 64): block tile 128(M) x 256(N)
    gemm_bf16_wt<true><<<gg, 256, 0, stream>>>(xb, wqb, qb, kM, kC, kC);
    gemm_bf16_wt<true><<<gg, 256, 0, stream>>>(xb, wkb, kb, kM, kC, kC);
    gemm_bf16_wt<true><<<gg, 256, 0, stream>>>(xb, wvb, vb, kM, kC, kC);

    const long nPairs = (long)kB * kT * kH * (kD / 2);   // 8,388,608
    rope_kernel<<<(unsigned)((nPairs + 255) / 256), 256, 0, stream>>>(qb, kb, fcos, fsin);

    attn_kernel<<<dim3(kB * kH, kT / 64), 128, 0, stream>>>(qb, kb, vb, yb);

    gemm_bf16_wt<false><<<gg, 256, 0, stream>>>(yb, wob, d_out, kM, kC, kC);
}
